// Block_6777458393276
// MI455X (gfx1250) — matrix operand order repaired and checked by reading, not compiled, not run on hardware
//
#include <hip/hip_runtime.h>

#define BB 2
#define TT 1024
#define CC 1024
#define HH 16
#define DD 64
#define MMEM 256
#define SS (TT + 2*MMEM)   // 1536
#define CK 4
#define REGW 0.01f

typedef _Float16 v8h  __attribute__((ext_vector_type(8)));
typedef _Float16 v16h __attribute__((ext_vector_type(16)));
typedef float    v8f  __attribute__((ext_vector_type(8)));

__device__ __forceinline__ v16h cat8(v8h lo, v8h hi) {
  return __builtin_shufflevector(lo, hi, 0,1,2,3,4,5,6,7,8,9,10,11,12,13,14,15);
}
__device__ __forceinline__ v8f zero8() {
  v8f z;
  for (int i = 0; i < 8; ++i) z[i] = 0.f;
  return z;
}
__device__ __forceinline__ v8f wmma_f16(v16h a, v16h b, v8f c) {
  return __builtin_amdgcn_wmma_f32_16x16x32_f16(false, a, false, b, (short)0, c, false, false);
}
// CDNA5 async global->LDS copy (16B per lane), tracked by ASYNCcnt
__device__ __forceinline__ void async_copy_b128(unsigned lds_byte_off, const _Float16* gsrc) {
  asm volatile("global_load_async_to_lds_b128 %0, %1, off"
               :: "v"(lds_byte_off), "v"(gsrc) : "memory");
}
__device__ __forceinline__ void wait_async0() {
  asm volatile("s_wait_asynccnt 0x0" ::: "memory");
}

// ---- pack concat([x, fwd_mem, rev_mem]) -> f16 kv [B,S,C] ----
__global__ void pack_kv(const float* __restrict__ x, const float* __restrict__ fm,
                        const float* __restrict__ rm, _Float16* __restrict__ kv) {
  size_t idx = (size_t)blockIdx.x * blockDim.x + threadIdx.x;
  if (idx >= (size_t)BB*SS*CC) return;
  int c = (int)(idx % CC);
  int s = (int)((idx / CC) % SS);
  int b = (int)(idx / ((size_t)SS*CC));
  float v;
  if (s < TT)            v = x [((size_t)b*TT   + s)          *CC + c];
  else if (s < TT+MMEM)  v = fm[((size_t)b*MMEM + (s-TT))     *CC + c];
  else                   v = rm[((size_t)b*MMEM + (s-TT-MMEM))*CC + c];
  kv[idx] = (_Float16)v;
}

// ---- weight f32 [K,N] -> f16 transposed [N,K] (B-operand is N-major) ----
__global__ void wtrans(const float* __restrict__ W, _Float16* __restrict__ Wt) {
  size_t idx = (size_t)blockIdx.x * blockDim.x + threadIdx.x;
  if (idx >= (size_t)CC*CC) return;
  int n = (int)(idx % CC);
  int k = (int)(idx / CC);
  Wt[(size_t)n*CC + k] = (_Float16)W[(size_t)k*CC + n];
}

// ---- WMMA GEMM: 8-wave block computes 128x64 of out = A[M,C] @ Bt[N,C]^T ----
// B panel (64 cols x 32 k) staged per stage via async DMA into double-buffered LDS.
// mode 0: f32 row-major (final out); mode 1: f16 row-major; mode 2: f16 V-transpose scatter
__global__ __launch_bounds__(256) void gemm16(const _Float16* __restrict__ A,
                                              const _Float16* __restrict__ Bt,
                                              float* __restrict__ outF,
                                              _Float16* __restrict__ outH,
                                              int rpbo, int rpbi, int mode) {
  __shared__ _Float16 ldsB[2 * 64 * 40];          // 2 x (64 rows x 32 halves, stride 40)
  const int tid  = threadIdx.x;
  const int wave = tid >> 5;
  const int lane = tid & 31;
  const int hfl  = lane >> 4, nl = lane & 15;
  const int r0   = blockIdx.x * 128 + wave * 16;
  const int n0   = blockIdx.y * 64;

  // staging assignment: thread copies one 16B chunk per stage
  const int sr  = tid >> 2;                        // 0..63 : B row within panel
  const int sc8 = (tid & 3) * 8;                   // 0,8,16,24 halves
  const _Float16* gsrc_base = Bt + (size_t)(n0 + sr)*CC + sc8;
  const unsigned lds_base = (unsigned)(size_t)&ldsB[sr*40 + sc8];

  const int ra = r0 + nl;
  const size_t arow = (size_t)(ra / rpbo) * rpbi + (ra % rpbo);
  const _Float16* Ap = A + arow * CC;

  v8f acc[4];
#pragma unroll
  for (int j = 0; j < 4; ++j) acc[j] = zero8();

  async_copy_b128(lds_base, gsrc_base);            // stage 0 -> buf 0

  for (int st = 0; st < CC/32; ++st) {
    const int kb  = st * 32;
    const int buf = st & 1;
    wait_async0();
    __syncthreads();                               // current buffer ready for all waves
    if (st + 1 < CC/32)                            // overlap next DMA with this stage's math
      async_copy_b128(lds_base + (unsigned)(buf ^ 1) * 5120u, gsrc_base + kb + 32);

    v16h a = cat8(*(const v8h*)(Ap + kb + hfl*8),
                  *(const v8h*)(Ap + kb + 16 + hfl*8));
    const _Float16* lb = ldsB + buf * 2560;
    v16h bf[4];
#pragma unroll
    for (int j = 0; j < 4; ++j)
      bf[j] = cat8(*(const v8h*)(lb + (j*16 + nl)*40 + hfl*8),
                   *(const v8h*)(lb + (j*16 + nl)*40 + 16 + hfl*8));
#pragma unroll
    for (int j = 0; j < 4; ++j) acc[j] = wmma_f16(a, bf[j], acc[j]);
  }

#pragma unroll
  for (int j = 0; j < 4; ++j) {
#pragma unroll
    for (int v = 0; v < 8; ++v) {
      const int rr = r0 + v + 8*hfl;
      const int cc = n0 + j*16 + nl;
      const float val = acc[j][v];
      if (mode == 0)      outF[(size_t)rr*CC + cc] = val;
      else if (mode == 1) outH[(size_t)rr*CC + cc] = (_Float16)val;
      else {              // V: [B,S,C] -> Vt[b][c][s]
        const int b = rr / SS, s = rr % SS;
        outH[((size_t)b*CC + cc)*SS + s] = (_Float16)val;
      }
    }
  }
}

// ---- gate: g[b][h][t] = sigmoid(q . gate_W[:,h] + gate_b[h]) ----
__global__ void gate_kernel(const _Float16* __restrict__ Q16, const float* __restrict__ gW,
                            const float* __restrict__ gb, float* __restrict__ g) {
  int idx = blockIdx.x * blockDim.x + threadIdx.x;   // B*T*H
  int h = idx % HH;
  int t = (idx / HH) % TT;
  int b = idx / (HH*TT);
  float acc = gb[h];
  const _Float16* q = Q16 + ((size_t)b*TT + t)*CC;
  for (int c = 0; c < CC; ++c) acc += (float)q[c] * gW[c*HH + h];
  g[((size_t)b*HH + h)*TT + t] = 1.f / (1.f + __expf(-acc));
}

// ---- flash attention, one wave per (b, h, 16-query tile) ----
__global__ __launch_bounds__(32) void attn_kernel(const _Float16* __restrict__ Q16,
                                                  const _Float16* __restrict__ K16,
                                                  const _Float16* __restrict__ Vt16,
                                                  const float* __restrict__ gbuf,
                                                  float* __restrict__ AO) {
  __shared__ _Float16 ldsP[16 * 40];   // 16x32 P tile, padded stride 40
  const int lane = threadIdx.x;
  const int hfl = lane >> 4, nl = lane & 15;
  const int t0 = blockIdx.x * 16;
  const int h  = blockIdx.y;
  const int b  = blockIdx.z;

  // Q fragments (A operand), pre-scaled by 1/sqrt(D) = 0.125
  const _Float16* Qp = Q16 + ((size_t)b*TT + t0 + nl)*CC + h*DD;
  v16h q0 = cat8(*(const v8h*)(Qp + hfl*8),      *(const v8h*)(Qp + 16 + hfl*8));
  v16h q1 = cat8(*(const v8h*)(Qp + 32 + hfl*8), *(const v8h*)(Qp + 48 + hfl*8));
#pragma unroll
  for (int i = 0; i < 16; ++i) { q0[i] = q0[i] * (_Float16)0.125f; q1[i] = q1[i] * (_Float16)0.125f; }

  v8f accC[4], accM[4];
#pragma unroll
  for (int j = 0; j < 4; ++j) { accC[j] = zero8(); accM[j] = zero8(); }
  float mrun[8], lrun[8];
#pragma unroll
  for (int v = 0; v < 8; ++v) { mrun[v] = -1e30f; lrun[v] = 0.f; }

  const int chunk_hi = (((t0 + 15) >> 5) << 5) + 32;   // skip fully-masked causal blocks

#pragma unroll
  for (int phase = 0; phase < 2; ++phase) {
    const int klo = phase ? TT : 0;
    const int khi = phase ? SS : chunk_hi;
    for (int kb = klo; kb < khi; kb += 32) {
      // ---- preload all K fragments (keeps loads overlapped, avoids per-wmma waits) ----
      const _Float16* Kp0 = K16 + ((size_t)b*SS + kb + nl)*CC + h*DD;
      const _Float16* Kp1 = Kp0 + (size_t)16*CC;
      v16h bk00 = cat8(*(const v8h*)(Kp0 + hfl*8),      *(const v8h*)(Kp0 + 16 + hfl*8));
      v16h bk01 = cat8(*(const v8h*)(Kp0 + 32 + hfl*8), *(const v8h*)(Kp0 + 48 + hfl*8));
      v16h bk10 = cat8(*(const v8h*)(Kp1 + hfl*8),      *(const v8h*)(Kp1 + 16 + hfl*8));
      v16h bk11 = cat8(*(const v8h*)(Kp1 + 32 + hfl*8), *(const v8h*)(Kp1 + 48 + hfl*8));
      v8f sc[2];
      sc[0] = wmma_f16(q0, bk00, zero8());
      sc[0] = wmma_f16(q1, bk01, sc[0]);
      sc[1] = wmma_f16(q0, bk10, zero8());
      sc[1] = wmma_f16(q1, bk11, sc[1]);

      if (phase == 0) {
#pragma unroll
        for (int jt = 0; jt < 2; ++jt) {
          const int s = kb + jt*16 + nl;
#pragma unroll
          for (int v = 0; v < 8; ++v) {
            const int t = t0 + v + 8*hfl;
            if (s > t) sc[jt][v] = -1e30f;
          }
        }
      }
      // ---- online softmax stats ----
      float mnew[8], alpha[8];
#pragma unroll
      for (int v = 0; v < 8; ++v) {
        float mx = fmaxf(sc[0][v], sc[1][v]);
        for (int msk = 1; msk < 16; msk <<= 1) mx = fmaxf(mx, __shfl_xor(mx, msk, 32));
        mnew[v]  = fmaxf(mrun[v], mx);
        alpha[v] = __expf(mrun[v] - mnew[v]);
        mrun[v]  = mnew[v];
      }
#pragma unroll
      for (int jt = 0; jt < 2; ++jt) {
#pragma unroll
        for (int v = 0; v < 8; ++v) {
          const float p = __expf(sc[jt][v] - mnew[v]);
          sc[jt][v] = p;
          ldsP[(v + 8*hfl)*40 + jt*16 + nl] = (_Float16)p;   // D-layout -> row-major tile
        }
      }
#pragma unroll
      for (int v = 0; v < 8; ++v) {
        float r = sc[0][v] + sc[1][v];
        for (int msk = 1; msk < 16; msk <<= 1) r += __shfl_xor(r, msk, 32);
        lrun[v] = lrun[v]*alpha[v] + r;
      }
#pragma unroll
      for (int j = 0; j < 4; ++j) {
#pragma unroll
        for (int v = 0; v < 8; ++v) { accC[j][v] *= alpha[v]; accM[j][v] *= alpha[v]; }
      }
      __syncthreads();
      // re-load P in A-operand layout
      v16h pa = cat8(*(const v8h*)&ldsP[nl*40 + hfl*8],
                     *(const v8h*)&ldsP[nl*40 + 16 + hfl*8]);
      // ---- P @ V : preload all 4 V fragments, then 4 WMMAs ----
      v16h bv[4];
#pragma unroll
      for (int j = 0; j < 4; ++j) {
        const _Float16* Vp = Vt16 + ((size_t)b*CC + h*DD + j*16 + nl)*SS + kb + hfl*8;
        bv[j] = cat8(*(const v8h*)Vp, *(const v8h*)(Vp + 16));
      }
#pragma unroll
      for (int j = 0; j < 4; ++j) {
        if (phase == 0) accC[j] = wmma_f16(pa, bv[j], accC[j]);
        else            accM[j] = wmma_f16(pa, bv[j], accM[j]);
      }
      __syncthreads();
    }
  }
  // ---- epilogue: Y = (acc_chunk + g * acc_mem) / l ----
  float* AOp = AO + (size_t)b*TT*CC + h*DD;
#pragma unroll
  for (int v = 0; v < 8; ++v) {
    const int m = v + 8*hfl;
    const float g   = gbuf[((size_t)b*HH + h)*TT + t0 + m];
    const float inv = 1.f / lrun[v];
#pragma unroll
    for (int j = 0; j < 4; ++j)
      AOp[(size_t)(t0 + m)*CC + j*16 + nl] = (accC[j][v] + g*accM[j][v]) * inv;
  }
}

// ---- causal depthwise conv residual, emit f16 for final GEMM ----
__global__ void conv_kernel(const float* __restrict__ AO, const float* __restrict__ cw,
                            const float* __restrict__ cb, _Float16* __restrict__ A2h) {
  size_t idx = (size_t)blockIdx.x * blockDim.x + threadIdx.x;
  if (idx >= (size_t)BB*TT*CC) return;
  int c = (int)(idx % CC);
  int t = (int)((idx / CC) % TT);
  int b = (int)(idx / ((size_t)TT*CC));
  const float* base = AO + (size_t)b*TT*CC + c;
  float acc = cb[c];
#pragma unroll
  for (int j = 0; j < CK; ++j) {
    const int tt = t - (CK - 1) + j;
    if (tt >= 0) acc += cw[c*CK + j] * base[(size_t)tt*CC];
  }
  A2h[idx] = (_Float16)(AO[idx] + acc);
}

// ---- reg_loss = REG * mean(|g|) ----
__global__ void reg_kernel(const float* __restrict__ gbuf, float* __restrict__ out) {
  __shared__ float red[256];
  float s = 0.f;
  for (int i = threadIdx.x; i < BB*HH*TT; i += 256) s += fabsf(gbuf[i]);
  red[threadIdx.x] = s;
  __syncthreads();
  for (int st = 128; st > 0; st >>= 1) {
    if ((int)threadIdx.x < st) red[threadIdx.x] += red[threadIdx.x + st];
    __syncthreads();
  }
  if (threadIdx.x == 0) out[0] = REGW * red[0] / (float)(BB*HH*TT);
}

extern "C" void kernel_launch(void* const* d_in, const int* in_sizes, int n_in,
                              void* d_out, int out_size, void* d_ws, size_t ws_size,
                              hipStream_t stream) {
  const float* x  = (const float*)d_in[0];
  const float* fm = (const float*)d_in[1];
  const float* rm = (const float*)d_in[2];
  const float* Wq = (const float*)d_in[3];
  const float* Wk = (const float*)d_in[4];
  const float* Wv = (const float*)d_in[5];
  const float* Wo = (const float*)d_in[6];
  const float* gW = (const float*)d_in[7];
  const float* gb = (const float*)d_in[8];
  const float* cw = (const float*)d_in[9];
  const float* cb = (const float*)d_in[10];
  float* out = (float*)d_out;

  _Float16* p = (_Float16*)d_ws;
  _Float16* kv16 = p; p += (size_t)BB*SS*CC;
  _Float16* Wqt  = p; p += (size_t)CC*CC;
  _Float16* Wkt  = p; p += (size_t)CC*CC;
  _Float16* Wvt  = p; p += (size_t)CC*CC;
  _Float16* Wot  = p; p += (size_t)CC*CC;
  _Float16* Q16  = p; p += (size_t)BB*TT*CC;
  _Float16* K16  = p; p += (size_t)BB*SS*CC;
  _Float16* Vt16 = p; p += (size_t)BB*CC*SS;
  _Float16* A2h  = p; p += (size_t)BB*TT*CC;
  float* gbuf = (float*)p;
  float* AO   = gbuf + (size_t)BB*HH*TT;

  pack_kv<<<(int)(((size_t)BB*SS*CC + 255)/256), 256, 0, stream>>>(x, fm, rm, kv16);
  wtrans<<<(int)(((size_t)CC*CC + 255)/256), 256, 0, stream>>>(Wq, Wqt);
  wtrans<<<(int)(((size_t)CC*CC + 255)/256), 256, 0, stream>>>(Wk, Wkt);
  wtrans<<<(int)(((size_t)CC*CC + 255)/256), 256, 0, stream>>>(Wv, Wvt);
  wtrans<<<(int)(((size_t)CC*CC + 255)/256), 256, 0, stream>>>(Wo, Wot);

  dim3 gq((BB*TT)/128, CC/64);   // 16 x 16
  dim3 gk((BB*SS)/128, CC/64);   // 24 x 16
  gemm16<<<gq, 256, 0, stream>>>(kv16, Wqt, nullptr, Q16,  TT, SS, 1);
  gemm16<<<gk, 256, 0, stream>>>(kv16, Wkt, nullptr, K16,  SS, SS, 1);
  gemm16<<<gk, 256, 0, stream>>>(kv16, Wvt, nullptr, Vt16, SS, SS, 2);

  gate_kernel<<<(BB*TT*HH)/256, 256, 0, stream>>>(Q16, gW, gb, gbuf);

  dim3 ga(TT/16, HH, BB);
  attn_kernel<<<ga, 32, 0, stream>>>(Q16, K16, Vt16, gbuf, AO);

  conv_kernel<<<(int)(((size_t)BB*TT*CC + 255)/256), 256, 0, stream>>>(AO, cw, cb, A2h);

  gemm16<<<gq, 256, 0, stream>>>(A2h, Wot, out, nullptr, TT, TT, 0);

  reg_kernel<<<1, 256, 0, stream>>>(gbuf, out + (size_t)BB*TT*CC);
}
